// ECGCNN_MoE_1005022347831
// MI455X (gfx1250) — matrix operand order repaired and checked by reading, not compiled, not run on hardware
//
#include <hip/hip_runtime.h>
#include <math.h>

typedef __bf16 bf16;
typedef __attribute__((ext_vector_type(16))) __bf16 v16bf;
typedef __attribute__((ext_vector_type(8)))  __bf16 v8bf;
typedef __attribute__((ext_vector_type(8)))  float  v8f;

#define NEXP 8
#define NB   64

// ---------------------------------------------------------------------------
// utility kernels
// ---------------------------------------------------------------------------
__global__ void zero_f32_kernel(float* __restrict__ p, size_t n) {
  size_t i = (size_t)blockIdx.x * blockDim.x + threadIdx.x;
  size_t stride = (size_t)gridDim.x * blockDim.x;
  for (; i < n; i += stride) p[i] = 0.0f;
}

__global__ void f2bf_kernel(const float* __restrict__ in, bf16* __restrict__ out, size_t n) {
  size_t i = (size_t)blockIdx.x * blockDim.x + threadIdx.x;
  size_t stride = (size_t)gridDim.x * blockDim.x;
  for (; i < n; i += stride) out[i] = (bf16)in[i];
}

// layer-1 weight repack: [E,64,16,3] fp32 -> [E,64,16*4] bf16, 4th tap zero.
// Pads layer-1 GEMM-K from 48 to 64 so every layer has K % 32 == 0.
__global__ void padw1_kernel(const float* __restrict__ in, bf16* __restrict__ out) {
  int i = blockIdx.x * blockDim.x + threadIdx.x;  // over E*64*16*4 = 32768
  if (i >= NEXP * 64 * 16 * 4) return;
  int kk = i & 3;
  int rest = i >> 2;  // (e*64+co)*16+ci
  out[i] = (kk < 3) ? (bf16)in[rest * 3 + kk] : (bf16)0.0f;
}

// fold conv-bias + BN into per-channel scale/shift: y = sc*x + sh, once per launch
__global__ void bnprep_kernel(const float* __restrict__ eb, const float* __restrict__ eg,
                              const float* __restrict__ ebb, const float* __restrict__ em,
                              const float* __restrict__ ev, float2* __restrict__ scsh, int n) {
  int i = blockIdx.x * blockDim.x + threadIdx.x;
  if (i >= n) return;
  float s = eg[i] * rsqrtf(ev[i] + 1e-5f);
  scsh[i] = make_float2(s, s * (eb[i] - em[i]) + ebb[i]);
}

// ---------------------------------------------------------------------------
// stem: conv1 (1->16, k=3, pad 1) + ReLU ; writes bf16 activations + fp32 mean pool
// ---------------------------------------------------------------------------
__global__ __launch_bounds__(256)
void stem_kernel(const float* __restrict__ x, const float* __restrict__ w,
                 const float* __restrict__ bc, bf16* __restrict__ hbf,
                 float* __restrict__ pooled) {
  __shared__ float xs[514];
  __shared__ float wl[48];
  __shared__ float bl[16];
  __shared__ float red[16];
  const int b = blockIdx.x;
  const int t = threadIdx.x;
  for (int i = t; i < 514; i += 256) {
    int l = i - 1;
    xs[i] = (l >= 0 && l < 512) ? x[b * 512 + l] : 0.0f;
  }
  if (t < 48) wl[t] = w[t];
  if (t < 16) { bl[t] = bc[t]; red[t] = 0.0f; }
  __syncthreads();
  float ps[16];
#pragma unroll
  for (int c = 0; c < 16; ++c) ps[c] = 0.0f;
  for (int l = t; l < 512; l += 256) {
    float x0 = xs[l], x1 = xs[l + 1], x2 = xs[l + 2];
#pragma unroll
    for (int c = 0; c < 16; ++c) {
      float y = fmaxf(wl[c * 3] * x0 + wl[c * 3 + 1] * x1 + wl[c * 3 + 2] * x2 + bl[c], 0.0f);
      hbf[((size_t)b * 16 + c) * 512 + l] = (bf16)y;
      ps[c] += y;
    }
  }
#pragma unroll
  for (int c = 0; c < 16; ++c) atomicAdd(&red[c], ps[c]);
  __syncthreads();
  if (t < 16) pooled[b * 16 + t] = red[t] * (1.0f / 512.0f);
}

// ---------------------------------------------------------------------------
// router: logits -> softmax -> top2 -> normalized dense combine weights + cv^2
// ---------------------------------------------------------------------------
__global__ __launch_bounds__(64)
void router_kernel(const float* __restrict__ pooled, const float* __restrict__ rw,
                   const float* __restrict__ rb, float* __restrict__ wcomb,
                   float* __restrict__ cv_out) {
  __shared__ float mp[NEXP];
  const int b = threadIdx.x;
  if (b < NEXP) mp[b] = 0.0f;
  __syncthreads();
  float logits[NEXP];
  float mx = -1e30f;
#pragma unroll
  for (int e = 0; e < NEXP; ++e) {
    float s = rb[e];
#pragma unroll
    for (int c = 0; c < 16; ++c) s += pooled[b * 16 + c] * rw[e * 16 + c];
    logits[e] = s;
    mx = fmaxf(mx, s);
  }
  float p[NEXP];
  float sum = 0.0f;
#pragma unroll
  for (int e = 0; e < NEXP; ++e) { p[e] = expf(logits[e] - mx); sum += p[e]; }
#pragma unroll
  for (int e = 0; e < NEXP; ++e) p[e] /= sum;
  int i1 = 0; float v1 = p[0];
#pragma unroll
  for (int e = 1; e < NEXP; ++e) if (p[e] > v1) { v1 = p[e]; i1 = e; }
  int i2 = -1; float v2 = -1.0f;
#pragma unroll
  for (int e = 0; e < NEXP; ++e) {
    if (e == i1) continue;
    if (p[e] > v2) { v2 = p[e]; i2 = e; }
  }
  const float gs = v1 + v2;
#pragma unroll
  for (int e = 0; e < NEXP; ++e)
    wcomb[b * NEXP + e] = (e == i1) ? (v1 / gs) : ((e == i2) ? (v2 / gs) : 0.0f);
#pragma unroll
  for (int e = 0; e < NEXP; ++e) atomicAdd(&mp[e], p[e]);
  __syncthreads();
  if (b == 0) {
    float mean = 0.0f;
#pragma unroll
    for (int e = 0; e < NEXP; ++e) mean += mp[e] * (1.0f / 64.0f);
    mean *= (1.0f / NEXP);
    float var = 0.0f;
#pragma unroll
    for (int e = 0; e < NEXP; ++e) {
      float d = mp[e] * (1.0f / 64.0f) - mean;
      var += d * d;
    }
    var *= (1.0f / (NEXP - 1));
    float cv = sqrtf(var) / (mean + 1e-10f);
    cv_out[0] = cv * cv;
  }
}

// ---------------------------------------------------------------------------
// expert conv block as implicit GEMM via v_wmma_f32_16x16x32_bf16.
//   M = Co (MT tiles of 16 per wave, 4 waves/block), N = 16 output positions of
//   one sample, K = Ci*KW (multiple of 32 everywhere; layer 1 KW padded to 4).
//   One im2col B gather per chunk feeds MT WMMAs (B is identical across co-tiles).
//   Input window async-staged into LDS (16B-aligned 8-elem groups; LIN and clamp
//   points are multiples of 8 so groups are fully in-bounds or fully zero).
//   Epilogue: prefolded BN scale/shift (float2), ReLU, maxpool2 via shfl_xor;
//   pool-pair lanes split the 8 channel rows (even lane: r0-3, odd: r4-7) so all
//   lanes store with no EXEC predication.
// ---------------------------------------------------------------------------
template <int CI, int CO, int KW, int PAD, int LIN, int MT, bool LAST>
__global__ __launch_bounds__(128)
void expert_layer_kernel(const bf16* __restrict__ Xin, long long in_estride,
                         const bf16* __restrict__ Wc, const float2* __restrict__ scsh,
                         const float* __restrict__ wcomb, bf16* __restrict__ Xout,
                         float* __restrict__ out_acc) {
  constexpr int LEFT = ((PAD + 7) / 8) * 8;          // aligned left halo
  constexpr int SH   = LEFT - PAD;                   // read-index shift
  constexpr int W8   = ((SH + 15 + KW + 7) / 8) * 8; // aligned row stride (elems)
  constexpr int K    = CI * KW;                      // GEMM K extent (K%32==0)
  constexpr int KC   = K >> 5;                       // K chunks of 32
  constexpr int LT   = LIN / 16;                     // N tiles per sample
  constexpr int NGRP = CI * (W8 / 8);                // 8-element groups in window

  const int e = blockIdx.z;
  const int b = blockIdx.y;
  const float gate = wcomb[b * NEXP + e];
  if (gate == 0.0f) return;  // top-2 sparsity: wave-uniform skip

  const int lt = blockIdx.x % LT;
  const int cog = blockIdx.x / LT;
  const int wave = threadIdx.x >> 5;
  const int lane = threadIdx.x & 31;
  const int tile0 = cog * (4 * MT) + wave * MT;      // first co-tile of this wave

  __shared__ bf16 Xs[CI * W8];

  // ---- async staging of the input window into LDS ----
  {
    v8bf z;
#pragma unroll
    for (int j = 0; j < 8; ++j) z[j] = (bf16)0.0f;
    const int base0 = lt * 16 - LEFT;  // multiple of 8
    const bf16* src_base = Xin + in_estride * e + (size_t)b * CI * LIN;
    for (int gidx = threadIdx.x; gidx < NGRP; gidx += 128) {
      const int ci = gidx / (W8 / 8);
      const int g  = gidx - ci * (W8 / 8);
      const int l0 = base0 + g * 8;
      bf16* dst = &Xs[ci * W8 + g * 8];
      if (l0 >= 0 && l0 < LIN) {
        const bf16* src = src_base + (size_t)ci * LIN + l0;
        const unsigned ldsoff = (unsigned)(unsigned long long)dst;  // LDS byte offset
        asm volatile("global_load_async_to_lds_b128 %0, %1, off"
                     :: "v"(ldsoff), "v"(src) : "memory");
      } else {
        *(v8bf*)dst = z;  // zero padding (16B-aligned vector LDS store)
      }
    }
  }
  asm volatile("s_wait_asynccnt 0" ::: "memory");
  __syncthreads();

  const int kh = lane >> 4;  // lane half selects K sub-range
  const int n  = lane & 15;  // N column / A row within tile
  const bf16* Arow[MT];
#pragma unroll
  for (int t = 0; t < MT; ++t)
    Arow[t] = Wc + ((size_t)e * CO + (size_t)(tile0 + t) * 16 + n) * K;

  v8f acc[MT];
#pragma unroll
  for (int t = 0; t < MT; ++t)
#pragma unroll
    for (int j = 0; j < 8; ++j) acc[t][j] = 0.0f;

  for (int kc = 0; kc < KC; ++kc) {
    const int kbase = kc * 32;
    // ---- B fragment: im2col gather from LDS; one div per chunk, then
    //      incremental (kk, addr) advance with branchless wrap. Shared by MT WMMAs.
    v16bf bm;
    {
      const int k0 = kbase + kh * 16;
      const int ci0 = k0 / KW;  // one magic-div per chunk
      int kk = k0 - ci0 * KW;
      int addr = ci0 * W8 + SH + n + kk;
#pragma unroll
      for (int j = 0; j < 16; ++j) {
        bm[j] = Xs[addr];
        ++kk; ++addr;
        if (kk == KW) { kk = 0; addr += W8 - KW; }
      }
    }
    if (kc + 1 < KC) __builtin_prefetch(Arow[0] + kbase + 32, 0, 1);
    // ---- MT co-tiles: A is cheap contiguous 16B loads; B amortized MT x ----
#pragma unroll
    for (int t = 0; t < MT; ++t) {
      union { v16bf v; v8bf h[2]; } a;
      a.h[0] = *(const v8bf*)(Arow[t] + kbase + kh * 16);
      a.h[1] = *(const v8bf*)(Arow[t] + kbase + kh * 16 + 8);
      acc[t] = __builtin_amdgcn_wmma_f32_16x16x32_bf16(
          false, a.v, false, bm, (short)0, acc[t], false, false);
    }
  }

  // ---- epilogue: prefolded BN, ReLU, maxpool2 via lane-pair shuffle.
  //      Both lanes of a pool pair hold the pooled value and share lout, so the
  //      pair splits the 8 channel rows: even lane stores r0-3, odd lane r4-7.
  //      All lanes store -> no EXEC predication; one writer per (co,lout). ----
  const int l = lt * 16 + n;
  const int lout = l >> 1;             // identical for both lanes of a pair
  const int rbase = (lane & 1) * 4;    // row split within the pair
  const float2* pp = scsh + (size_t)e * CO;
#pragma unroll
  for (int t = 0; t < MT; ++t) {
    const int cobase = (tile0 + t) * 16;
    float pm[8];
#pragma unroll
    for (int r = 0; r < 8; ++r) {
      const float2 ss = pp[cobase + r + 8 * kh];
      float y = fmaxf(acc[t][r] * ss.x + ss.y, 0.0f);
      float yo = __shfl_xor(y, 1, 32);
      pm[r] = fmaxf(y, yo);
    }
#pragma unroll
    for (int q = 0; q < 4; ++q) {
      const int r = rbase + q;
      const int co = cobase + r + 8 * kh;
      if (LAST) {
        atomicAdd(&out_acc[((size_t)b * CO + co) * (LIN / 2) + lout], gate * pm[r]);
      } else {
        Xout[(((size_t)e * NB + b) * CO + co) * (LIN / 2) + lout] = (bf16)pm[r];
      }
    }
  }
}

// ---------------------------------------------------------------------------
// head: global mean over L=32, fc1(512->256)+ReLU, fc2(256->64)+ReLU, fc3(64->5)
// ---------------------------------------------------------------------------
__global__ __launch_bounds__(256)
void head_kernel(const float* __restrict__ outacc, const float* __restrict__ f1w,
                 const float* __restrict__ f1b, const float* __restrict__ f2w,
                 const float* __restrict__ f2b, const float* __restrict__ f3w,
                 const float* __restrict__ f3b, float* __restrict__ out) {
  __shared__ float gm[512];
  __shared__ float g1[256];
  __shared__ float g2[64];
  const int b = blockIdx.x;
  const int t = threadIdx.x;
  for (int c = t; c < 512; c += 256) {
    const float* p = outacc + ((size_t)b * 512 + c) * 32;
    float s = 0.0f;
#pragma unroll
    for (int i = 0; i < 32; ++i) s += p[i];
    gm[c] = s * (1.0f / 32.0f);
  }
  __syncthreads();
  {
    float s = f1b[t];
    const float* w = f1w + (size_t)t * 512;
    for (int k = 0; k < 512; ++k) s += w[k] * gm[k];
    g1[t] = fmaxf(s, 0.0f);
  }
  __syncthreads();
  if (t < 64) {
    float s = f2b[t];
    const float* w = f2w + (size_t)t * 256;
    for (int k = 0; k < 256; ++k) s += w[k] * g1[k];
    g2[t] = fmaxf(s, 0.0f);
  }
  __syncthreads();
  if (t < 5) {
    float s = f3b[t];
    const float* w = f3w + (size_t)t * 64;
#pragma unroll
    for (int k = 0; k < 64; ++k) s += w[k] * g2[k];
    out[b * 5 + t] = s;
  }
}

// ---------------------------------------------------------------------------
// host launcher
// ---------------------------------------------------------------------------
extern "C" void kernel_launch(void* const* d_in, const int* in_sizes, int n_in,
                              void* d_out, int out_size, void* d_ws, size_t ws_size,
                              hipStream_t stream) {
  (void)in_sizes; (void)n_in; (void)out_size; (void)ws_size;
  const float* x   = (const float*)d_in[0];
  const float* c1w = (const float*)d_in[1];
  const float* c1b = (const float*)d_in[2];
  const float *ew[4], *ebi[4], *egi[4], *ebbi[4], *emi[4], *evi[4];
  for (int i = 0; i < 4; ++i) {
    ew[i]   = (const float*)d_in[3 + i * 6];
    ebi[i]  = (const float*)d_in[4 + i * 6];
    egi[i]  = (const float*)d_in[5 + i * 6];
    ebbi[i] = (const float*)d_in[6 + i * 6];
    emi[i]  = (const float*)d_in[7 + i * 6];
    evi[i]  = (const float*)d_in[8 + i * 6];
  }
  const float* rw  = (const float*)d_in[27];
  const float* rb  = (const float*)d_in[28];
  const float* f1w = (const float*)d_in[29];
  const float* f1b = (const float*)d_in[30];
  const float* f2w = (const float*)d_in[31];
  const float* f2b = (const float*)d_in[32];
  const float* f3w = (const float*)d_in[33];
  const float* f3b = (const float*)d_in[34];
  float* out = (float*)d_out;

  char* ws = (char*)d_ws;
  size_t off = 0;
  auto carve = [&](size_t bytes) -> char* {
    char* p = ws + off;
    off = (off + bytes + 255) & ~(size_t)255;
    return p;
  };
  float* wcomb  = (float*)carve(NB * NEXP * 4);
  float* pooled = (float*)carve(NB * 16 * 4);                 // 4096 B (256-aligned)
  float* outacc = (float*)carve((size_t)NB * 512 * 32 * 4);   // contiguous with pooled
  bf16* hbf = (bf16*)carve((size_t)NB * 16 * 512 * 2);
  bf16* wc1 = (bf16*)carve((size_t)32768 * 2);                // padded layer-1 weights
  bf16* wc2 = (bf16*)carve((size_t)589824 * 2);
  bf16* wc3 = (bf16*)carve((size_t)2883584 * 2);
  bf16* wc4 = (bf16*)carve((size_t)26214400 * 2);
  bf16* a1  = (bf16*)carve((size_t)NEXP * NB * 64 * 256 * 2);
  bf16* a2  = (bf16*)carve((size_t)NEXP * NB * 128 * 128 * 2);
  bf16* a3  = (bf16*)carve((size_t)NEXP * NB * 256 * 64 * 2);
  float2* bn1 = (float2*)carve((size_t)NEXP * 64 * 8);
  float2* bn2 = (float2*)carve((size_t)NEXP * 128 * 8);
  float2* bn3 = (float2*)carve((size_t)NEXP * 256 * 8);
  float2* bn4 = (float2*)carve((size_t)NEXP * 512 * 8);

  // zero pooled + outacc (contiguous region)
  zero_f32_kernel<<<512, 256, 0, stream>>>(pooled, (size_t)NB * 16 + (size_t)NB * 512 * 32);

  // weight fp32 -> bf16 conversion (~60 MB, bandwidth-trivial at 23.3 TB/s)
  padw1_kernel<<<128, 256, 0, stream>>>(ew[0], wc1);
  f2bf_kernel<<<1024, 256, 0, stream>>>(ew[1], wc2, (size_t)589824);
  f2bf_kernel<<<2048, 256, 0, stream>>>(ew[2], wc3, (size_t)2883584);
  f2bf_kernel<<<4096, 256, 0, stream>>>(ew[3], wc4, (size_t)26214400);

  // fold conv-bias + BN into per-channel (scale, shift)
  bnprep_kernel<<<2, 256, 0, stream>>>(ebi[0], egi[0], ebbi[0], emi[0], evi[0], bn1, NEXP * 64);
  bnprep_kernel<<<4, 256, 0, stream>>>(ebi[1], egi[1], ebbi[1], emi[1], evi[1], bn2, NEXP * 128);
  bnprep_kernel<<<8, 256, 0, stream>>>(ebi[2], egi[2], ebbi[2], emi[2], evi[2], bn3, NEXP * 256);
  bnprep_kernel<<<16, 256, 0, stream>>>(ebi[3], egi[3], ebbi[3], emi[3], evi[3], bn4, NEXP * 512);

  stem_kernel<<<NB, 256, 0, stream>>>(x, c1w, c1b, hbf, pooled);
  router_kernel<<<1, NB, 0, stream>>>(pooled, rw, rb, wcomb, out + 320);

  // (co_groups * l_tiles, batch, expert); MT co-tiles per wave amortize B gather
  expert_layer_kernel<16, 64, 4, 1, 512, 1, false><<<dim3(32, NB, NEXP), 128, 0, stream>>>(
      hbf, 0LL, wc1, bn1, wcomb, a1, nullptr);
  expert_layer_kernel<64, 128, 9, 4, 256, 2, false><<<dim3(16, NB, NEXP), 128, 0, stream>>>(
      a1, (long long)NB * 64 * 256, wc2, bn2, wcomb, a2, nullptr);
  expert_layer_kernel<128, 256, 11, 5, 128, 4, false><<<dim3(8, NB, NEXP), 128, 0, stream>>>(
      a2, (long long)NB * 128 * 128, wc3, bn3, wcomb, a3, nullptr);
  expert_layer_kernel<256, 512, 25, 12, 64, 4, true><<<dim3(8, NB, NEXP), 128, 0, stream>>>(
      a3, (long long)NB * 256 * 64, wc4, bn4, wcomb, nullptr, outacc);

  head_kernel<<<NB, 256, 0, stream>>>(outacc, f1w, f1b, f2w, f2b, f3w, f3b, out);
}
